// TransformerBlock_59811714564188
// MI455X (gfx1250) — matrix operand order repaired and checked by reading, not compiled, not run on hardware
//
#include <hip/hip_runtime.h>
#include <hip/hip_bf16.h>
#include <math.h>

typedef _Float16 h16;
typedef __attribute__((ext_vector_type(16))) _Float16 v16h;
typedef __attribute__((ext_vector_type(8)))  _Float16 v8h;
typedef __attribute__((ext_vector_type(8)))  float    v8f;

#define DEVI static __device__ __forceinline__

// ---- WMMA fragment helpers (f16 16x16x32, per ISA 7.12.2 layouts) ----------
// A (16xK=32 tile), row-major storage with leading dim `ld` (halfs):
//   lane L: row = L&15; halves 0..7 = K[koff..koff+7], 8..15 = K[koff+16..koff+23]
//   koff = 0 for lanes 0-15, 8 for lanes 16-31.
DEVI v16h frag_a(const h16* p0, int ld) {
  const int lane = threadIdx.x & 31;
  const h16* p = p0 + (lane & 15) * ld + ((lane & 16) ? 8 : 0);
  union { v16h v; v8h h[2]; } u;
  u.h[0] = *(const v8h*)(p);
  u.h[1] = *(const v8h*)(p + 16);
  return u.v;
}
// B operand read from B^T stored row-major [col][k] with leading dim `ld`:
//   lane L: col = L&15; halves 0..7 = K[koff..koff+7], 8..15 = K[koff+16..koff+23], koff = (L<16)?0:8.
DEVI v16h frag_bt(const h16* p0, int ld) {
  const int lane = threadIdx.x & 31;
  const h16* p = p0 + (lane & 15) * ld + ((lane & 16) ? 8 : 0);
  union { v16h v; v8h h[2]; } u;
  u.h[0] = *(const v8h*)(p);
  u.h[1] = *(const v8h*)(p + 16);
  return u.v;
}
DEVI v8f wmma_f16(v16h a, v16h b, v8f c) {
  return __builtin_amdgcn_wmma_f32_16x16x32_f16(false, a, false, b, (short)0, c,
                                                false, false);
}
// D/C tile (f32): lane L holds col = L&15, rows (base + r + ((L&16)?8:0)), r=0..7.

// ---------------------------------------------------------------------------
__global__ void k_cvt(const float* __restrict__ s, h16* __restrict__ d, int n) {
  int i = blockIdx.x * 256 + threadIdx.x;
  if (i < n) d[i] = (h16)s[i];
}

// ---------------------------------------------------------------------------
// Per-window fused: window partition + LN1 + pos-bias + QKV (WMMA) + SA + EA
// attention -> y (f16, [512 windows][64 tok][1536]).
// grid=512, block=256 (8 waves), dynamic LDS = 282624 B.
__global__ void __launch_bounds__(256)
k_attn(const float* __restrict__ x, const float* __restrict__ g1,
       const float* __restrict__ b1, const h16* __restrict__ wea,
       const h16* __restrict__ wsa, const float* __restrict__ rpb,
       h16* __restrict__ y) {
  extern __shared__ char smem[];
  h16*   sXsa = (h16*)(smem);            // [4][64][32] f16  (16 KB)
  h16*   sXea = (h16*)(smem + 16384);    // [4][64][32] f16  (16 KB)
  float* sRPB = (float*)(smem + 32768);  // [225][4] f32     (4 KB)
  h16*   sQKV = (h16*)(smem + 36864);    // 196608 B (phase-dependent layout)
  float* sS   = (float*)(smem + 233472); // 2x[64][64] f32   (32 KB)
  h16*   sP   = (h16*)(smem + 266240);   // 2x[64][64] f16   (16 KB)

  const int tid = threadIdx.x, wv = tid >> 5, lane = tid & 31;
  const int wq = blockIdx.x, b = wq >> 8, hh = (wq >> 4) & 15, ww = wq & 15;

  for (int i = tid; i < 900; i += 256) sRPB[i] = rpb[i];

  // ---- window partition + LN1 + sine pos bias (thread = (exposure, token))
  {
    const int e = tid >> 6, n = tid & 63;
    const int Cidx = n >> 1, r0 = (n & 1) * 4;
    const float* xb = x + (size_t)((b * 4 + e) * 32 + Cidx) * 16384 +
                      (hh * 8) * 128 + ww * 8;
    float v[32];
    for (int rr = 0; rr < 4; ++rr) {
      const float4 q0 = *(const float4*)(xb + (r0 + rr) * 128);
      const float4 q1 = *(const float4*)(xb + (r0 + rr) * 128 + 4);
      v[rr * 8 + 0] = q0.x; v[rr * 8 + 1] = q0.y;
      v[rr * 8 + 2] = q0.z; v[rr * 8 + 3] = q0.w;
      v[rr * 8 + 4] = q1.x; v[rr * 8 + 5] = q1.y;
      v[rr * 8 + 6] = q1.z; v[rr * 8 + 7] = q1.w;
    }
    float mu = 0.f;
    for (int c = 0; c < 32; ++c) mu += v[c];
    mu *= 0.03125f;
    float var = 0.f;
    for (int c = 0; c < 32; ++c) { float d = v[c] - mu; var += d * d; }
    const float rstd = rsqrtf(var * 0.03125f + 1e-5f);
    const int i8 = n >> 3, j8 = n & 7;
    const float yv = (float)(i8 + 1) * 6.28318530718f / 8.000001f;
    const float xv = (float)(j8 + 1) * 6.28318530718f / 8.000001f;
    for (int c = 0; c < 32; ++c) {
      const float ln = (v[c] - mu) * rstd * g1[c] + b1[c];
      sXsa[(e * 64 + n) * 32 + c] = (h16)ln;
      const float base = (c < 16) ? yv : xv;
      const int m = ((c < 16) ? c : (c - 16)) >> 1;
      const float arg = base * powf(10000.f, -(float)m * 0.125f);
      const float pe = (c & 1) ? cosf(arg) : sinf(arg);
      sXea[(e * 64 + n) * 32 + c] = (h16)(ln + pe);
    }
  }
  __syncthreads();

  // ========================= PHASE A: self-attention =========================
  // Projection: [256 rows (e,n)] x [K=32] @ w_sa^T -> 384 cols (p,hd,d).
  // LDS layout: Q [hd][64][128] @0, K @32768, V^T [hd][128][64] @65536 (halfs).
  for (int t = wv; t < 384; t += 8) {
    const int mt = t / 24, nt = t % 24;
    v8f acc = {0.f, 0.f, 0.f, 0.f, 0.f, 0.f, 0.f, 0.f};
    acc = wmma_f16(frag_a(sXsa + mt * 16 * 32, 32),
                   frag_bt(wsa + nt * 16 * 32, 32), acc);
    // p/hd are constant over the 16-col tile; branchless base/stride select.
    const int p = nt >> 3, hd = (nt >> 1) & 3;
    const int d = (nt & 1) * 16 + (lane & 15);
    const int e = mt >> 2, nb = (mt & 3) * 16 + ((lane & 16) ? 8 : 0);
    const int baseQK = p * 32768 + hd * 8192 + nb * 128 + e * 32 + d;
    const int baseV  = 65536 + hd * 8192 + (e * 32 + d) * 64 + nb;
    const bool isV   = (p == 2);
    const int base   = isV ? baseV : baseQK;
    const int stride = isV ? 1 : 128;
    for (int r = 0; r < 8; ++r) sQKV[base + r * stride] = (h16)acc[r];
  }
  __syncthreads();

  for (int hp = 0; hp < 2; ++hp) {  // two heads at a time (S is 32 KB f32)
    // S = scale*Q@K^T + rel_pos_bias
    for (int t = wv; t < 32; t += 8) {
      const int hl = t >> 4, hd = hp * 2 + hl, it = (t >> 2) & 3, jt = t & 3;
      const h16* Qb = sQKV + hd * 8192 + it * 16 * 128;
      const h16* Kb = sQKV + 32768 + hd * 8192 + jt * 16 * 128;
      v8f acc = {0.f, 0.f, 0.f, 0.f, 0.f, 0.f, 0.f, 0.f};
      for (int ks = 0; ks < 4; ++ks)
        acc = wmma_f16(frag_a(Qb + ks * 32, 128), frag_bt(Kb + ks * 32, 128), acc);
      const int j = jt * 16 + (lane & 15), yj = j >> 3, xj = j & 7;
      const int ib = it * 16 + ((lane & 16) ? 8 : 0);
      for (int r = 0; r < 8; ++r) {
        const int i = ib + r, yi = i >> 3, xi = i & 7;
        const int ridx = (yi - yj + 7) * 15 + (xi - xj + 7);
        sS[hl * 4096 + i * 64 + j] = acc[r] * 0.1767766953f + sRPB[ridx * 4 + hd];
      }
    }
    __syncthreads();
    if (tid < 128) {  // softmax, thread = row
      const int hl = tid >> 6, n = tid & 63;
      const float* row = sS + hl * 4096 + n * 64;
      float mx = row[0];
      for (int j = 1; j < 64; ++j) mx = fmaxf(mx, row[j]);
      float sum = 0.f;
      for (int j = 0; j < 64; ++j) sum += expf(row[j] - mx);
      const float inv = 1.f / sum;
      for (int j = 0; j < 64; ++j)
        sP[hl * 4096 + n * 64 + j] = (h16)(expf(row[j] - mx) * inv);
    }
    __syncthreads();
    // O = P @ V  (V stored transposed -> contiguous B frags)
    for (int t = wv; t < 64; t += 8) {
      const int hl = t >> 5, hd = hp * 2 + hl, rem = t & 31;
      const int it = rem >> 3, dt = rem & 7;
      const h16* Pb = sP + hl * 4096 + it * 16 * 64;
      const h16* VTb = sQKV + 65536 + hd * 8192 + dt * 16 * 64;
      v8f acc = {0.f, 0.f, 0.f, 0.f, 0.f, 0.f, 0.f, 0.f};
      for (int ks = 0; ks < 2; ++ks)
        acc = wmma_f16(frag_a(Pb + ks * 32, 64), frag_bt(VTb + ks * 32, 64), acc);
      const int colg = hd * 128 + dt * 16 + (lane & 15);
      const int nb = it * 16 + ((lane & 16) ? 8 : 0);
      for (int r = 0; r < 8; ++r)
        y[(size_t)(wq * 64 + nb + r) * 1536 + colg] = (h16)acc[r];
    }
    __syncthreads();
  }

  // ===================== PHASE B: exposure-alignment =========================
  // Projection with w_ea on (LN+pos). LDS: Q/K [e][hd][64][32], V^T [e][hd][32][64].
  for (int t = wv; t < 384; t += 8) {
    const int mt = t / 24, nt = t % 24;
    v8f acc = {0.f, 0.f, 0.f, 0.f, 0.f, 0.f, 0.f, 0.f};
    acc = wmma_f16(frag_a(sXea + mt * 16 * 32, 32),
                   frag_bt(wea + nt * 16 * 32, 32), acc);
    const int p = nt >> 3, hd = (nt >> 1) & 3;
    const int d = (nt & 1) * 16 + (lane & 15);
    const int e = mt >> 2, nb = (mt & 3) * 16 + ((lane & 16) ? 8 : 0);
    const int baseQK = p * 32768 + e * 8192 + hd * 2048 + nb * 32 + d;
    const int baseV  = 65536 + e * 8192 + hd * 2048 + d * 64 + nb;
    const bool isV   = (p == 2);
    const int base   = isV ? baseV : baseQK;
    const int stride = isV ? 1 : 32;
    for (int r = 0; r < 8; ++r) sQKV[base + r * stride] = (h16)acc[r];
  }
  __syncthreads();

  // 32 attention instances: var 0 = ea1 (q=exp0, kv=e), var 1 = ea2 (q=e, kv=exp0)
  for (int it8 = 0; it8 < 16; ++it8) {  // two instances per iteration
    const int al = wv >> 2;
    const int aid = it8 * 2 + al;
    const int var = aid >> 4, e = (aid >> 2) & 3, hd = aid & 3;
    const int qe = var ? e : 0, ke = var ? 0 : e;
    {
      const h16* Qb = sQKV + qe * 8192 + hd * 2048;
      const h16* Kb = sQKV + 32768 + ke * 8192 + hd * 2048;
      for (int t = (wv & 3); t < 16; t += 4) {
        const int it = t >> 2, jt = t & 3;
        v8f acc = {0.f, 0.f, 0.f, 0.f, 0.f, 0.f, 0.f, 0.f};
        acc = wmma_f16(frag_a(Qb + it * 16 * 32, 32),
                       frag_bt(Kb + jt * 16 * 32, 32), acc);
        const int j = jt * 16 + (lane & 15);
        const int ib = it * 16 + ((lane & 16) ? 8 : 0);
        for (int r = 0; r < 8; ++r)
          sS[al * 4096 + (ib + r) * 64 + j] = acc[r] * 0.1767766953f;
      }
    }
    __syncthreads();
    if (tid < 128) {
      const int sl = tid >> 6, n = tid & 63;
      const float* row = sS + sl * 4096 + n * 64;
      float mx = row[0];
      for (int j = 1; j < 64; ++j) mx = fmaxf(mx, row[j]);
      float sum = 0.f;
      for (int j = 0; j < 64; ++j) sum += expf(row[j] - mx);
      const float inv = 1.f / sum;
      for (int j = 0; j < 64; ++j)
        sP[sl * 4096 + n * 64 + j] = (h16)(expf(row[j] - mx) * inv);
    }
    __syncthreads();
    {
      const h16* VTb = sQKV + 65536 + ke * 8192 + hd * 2048;
      for (int t = (wv & 3); t < 8; t += 4) {
        const int it = t >> 1, dt = t & 1;
        v8f acc = {0.f, 0.f, 0.f, 0.f, 0.f, 0.f, 0.f, 0.f};
        for (int ks = 0; ks < 2; ++ks)
          acc = wmma_f16(frag_a(sP + al * 4096 + it * 16 * 64 + ks * 32, 64),
                         frag_bt(VTb + dt * 16 * 64 + ks * 32, 64), acc);
        const int colg = (var + 1) * 512 + hd * 128 + e * 32 + dt * 16 + (lane & 15);
        const int nb = it * 16 + ((lane & 16) ? 8 : 0);
        for (int r = 0; r < 8; ++r)
          y[(size_t)(wq * 64 + nb + r) * 1536 + colg] = (h16)acc[r];
      }
    }
    __syncthreads();
  }
}

// ---------------------------------------------------------------------------
// Per-window GEGLU (1536->128) + fc2 + window reverse + residual -> x1.
// grid=512, block=256, dynamic LDS = 278528 B.
__global__ void __launch_bounds__(256)
k_geglu1(const h16* __restrict__ y, const h16* __restrict__ w11,
         const h16* __restrict__ w12, const float* __restrict__ b11,
         const float* __restrict__ b12, const h16* __restrict__ w2,
         const float* __restrict__ b2, const float* __restrict__ x,
         float* __restrict__ x1) {
  extern __shared__ char smem[];
  h16*   sY = (h16*)smem;                // [64][1536] f16 (192 KB)
  float* sH = (float*)(smem + 196608);   // [64][256] f32  (64 KB)
  h16*   sG = (h16*)(smem + 262144);     // [64][128] f16  (16 KB)

  const int tid = threadIdx.x, wv = tid >> 5, lane = tid & 31;
  const int wq = blockIdx.x, b = wq >> 8, hh = (wq >> 4) & 15, ww = wq & 15;

  // ---- async global->LDS bulk copy of the 192 KB y tile (ASYNCcnt path) ----
  {
    const char* src = (const char*)(y + (size_t)wq * 98304);
    const unsigned lds0 = (unsigned)(uintptr_t)sY;  // LDS byte address (low 32b)
    for (int i = tid; i < 12288; i += 256) {        // 48 full iterations, no divergence
      const unsigned dst = lds0 + (unsigned)i * 16u;
      const unsigned long long ga =
          (unsigned long long)(uintptr_t)(src + (size_t)i * 16);
      asm volatile("global_load_async_to_lds_b128 %0, %1, off"
                   :: "v"(dst), "v"(ga) : "memory");
    }
    asm volatile("s_wait_asynccnt 0x0" ::: "memory");
  }
  __syncthreads();

  // y @ [fc11^T | fc12^T]: 64x1536 x 1536x256, B frags streamed from L2 (f16)
  const int mt = wv >> 1, nbase = (wv & 1) * 8;
  // wave-uniform weight/bias selection hoisted out of the k-loop
  const h16* wsel   = (nbase == 0) ? w11 : w12;
  const float* bsel = (nbase == 0) ? b11 : b12;
  const int cbase   = nbase * 16;  // 0 or 128
  v8f acc[8];
  for (int i = 0; i < 8; ++i) acc[i] = (v8f){0.f, 0.f, 0.f, 0.f, 0.f, 0.f, 0.f, 0.f};
  for (int ks = 0; ks < 48; ++ks) {
    v16h a = frag_a(sY + mt * 16 * 1536 + ks * 32, 1536);
    for (int ni = 0; ni < 8; ++ni) {
      const h16* bt = wsel + (ni * 16) * 1536;  // (col0 - cbase) * ld
      acc[ni] = wmma_f16(a, frag_bt(bt + ks * 32, 1536), acc[ni]);
    }
  }
  for (int ni = 0; ni < 8; ++ni) {
    const int col = cbase + ni * 16 + (lane & 15);
    const float bias = bsel[ni * 16 + (lane & 15)];
    const int mb = mt * 16 + ((lane & 16) ? 8 : 0);
    for (int r = 0; r < 8; ++r) sH[(mb + r) * 256 + col] = acc[ni][r] + bias;
  }
  __syncthreads();
  for (int i = tid; i < 8192; i += 256) {  // exact-gelu gate
    const int n = i >> 7, o = i & 127;
    const float h1v = sH[n * 256 + o], h2v = sH[n * 256 + 128 + o];
    sG[i] = (h16)(0.5f * h1v * (1.f + erff(h1v * 0.70710678f)) * h2v);
  }
  __syncthreads();
  // g @ fc2^T (64x128 x 128x128) + window reverse + residual
  for (int t = wv; t < 32; t += 8) {
    const int m2t = t >> 3, dt2 = t & 7;
    v8f a2 = {0.f, 0.f, 0.f, 0.f, 0.f, 0.f, 0.f, 0.f};
    for (int ks = 0; ks < 4; ++ks)
      a2 = wmma_f16(frag_a(sG + m2t * 16 * 128 + ks * 32, 128),
                    frag_bt(w2 + dt2 * 16 * 128 + ks * 32, 128), a2);
    const int mcol = dt2 * 16 + (lane & 15);
    const int e = mcol >> 5, ch = ((mcol >> 3) & 3) * 8 + (mcol & 7);
    const float bias = b2[mcol];
    const int nb2 = m2t * 16 + ((lane & 16) ? 8 : 0);
    for (int r = 0; r < 8; ++r) {
      const int n = nb2 + r, wsr = n >> 3, wsc = n & 7;
      const size_t gi = (size_t)((b * 4 + e) * 32 + ch) * 16384 +
                        (wsr * 16 + hh) * 128 + (wsc * 16 + ww);
      x1[gi] = a2[r] + bias + x[gi];
    }
  }
}

// ---------------------------------------------------------------------------
// LN2 + GEGLU MLP (128->128) + final residual -> out.
// grid=512 (64 tokens each), block=256, dynamic LDS = 131072 B.
__global__ void __launch_bounds__(256)
k_mlp2(const float* __restrict__ x1, const float* __restrict__ g2,
       const float* __restrict__ bt2, const h16* __restrict__ w11,
       const float* __restrict__ b11, const h16* __restrict__ w12,
       const float* __restrict__ b12, const h16* __restrict__ w2,
       const float* __restrict__ b2, float* __restrict__ out) {
  extern __shared__ char smem[];
  float* sX = (float*)smem;              // [64][128] f32 (32 KB)
  h16*   sA = (h16*)(smem + 32768);      // [64][128] f16 (16 KB)
  float* sH = (float*)(smem + 49152);    // [64][256] f32 (64 KB)
  h16*   sG = (h16*)(smem + 114688);     // [64][128] f16 (16 KB)

  const int tid = threadIdx.x, wv = tid >> 5, lane = tid & 31;
  const int tok0 = blockIdx.x * 64;

  for (int i = tid; i < 8192; i += 256) {
    const int n = i >> 7, f = i & 127;
    const int flat = tok0 + n;
    sX[i] = x1[(size_t)((flat >> 14) * 128 + f) * 16384 + (flat & 16383)];
  }
  __syncthreads();
  if (tid < 64) {  // LN2 over 128 channels, thread = token
    float mu = 0.f;
    for (int f = 0; f < 128; ++f) mu += sX[tid * 128 + f];
    mu *= (1.f / 128.f);
    float var = 0.f;
    for (int f = 0; f < 128; ++f) { float d = sX[tid * 128 + f] - mu; var += d * d; }
    const float rstd = rsqrtf(var * (1.f / 128.f) + 1e-5f);
    for (int f = 0; f < 128; ++f)
      sA[tid * 128 + f] = (h16)((sX[tid * 128 + f] - mu) * rstd * g2[f] + bt2[f]);
  }
  __syncthreads();
  {
    const int mt = wv >> 1, nbase = (wv & 1) * 8;
    const h16* wsel   = (nbase == 0) ? w11 : w12;   // wave-uniform
    const float* bsel = (nbase == 0) ? b11 : b12;
    const int cbase   = nbase * 16;
    v8f acc[8];
    for (int i = 0; i < 8; ++i) acc[i] = (v8f){0.f, 0.f, 0.f, 0.f, 0.f, 0.f, 0.f, 0.f};
    for (int ks = 0; ks < 4; ++ks) {
      v16h a = frag_a(sA + mt * 16 * 128 + ks * 32, 128);
      for (int ni = 0; ni < 8; ++ni) {
        const h16* bt = wsel + (ni * 16) * 128;
        acc[ni] = wmma_f16(a, frag_bt(bt + ks * 32, 128), acc[ni]);
      }
    }
    for (int ni = 0; ni < 8; ++ni) {
      const int col = cbase + ni * 16 + (lane & 15);
      const float bias = bsel[ni * 16 + (lane & 15)];
      const int mb = mt * 16 + ((lane & 16) ? 8 : 0);
      for (int r = 0; r < 8; ++r) sH[(mb + r) * 256 + col] = acc[ni][r] + bias;
    }
  }
  __syncthreads();
  for (int i = tid; i < 8192; i += 256) {
    const int n = i >> 7, o = i & 127;
    const float h1v = sH[n * 256 + o], h2v = sH[n * 256 + 128 + o];
    sG[i] = (h16)(0.5f * h1v * (1.f + erff(h1v * 0.70710678f)) * h2v);
  }
  __syncthreads();
  for (int t = wv; t < 32; t += 8) {
    const int mt = t >> 3, dt = t & 7;
    v8f acc = {0.f, 0.f, 0.f, 0.f, 0.f, 0.f, 0.f, 0.f};
    for (int ks = 0; ks < 4; ++ks)
      acc = wmma_f16(frag_a(sG + mt * 16 * 128 + ks * 32, 128),
                     frag_bt(w2 + dt * 16 * 128 + ks * 32, 128), acc);
    const int fcol = dt * 16 + (lane & 15);
    const float bias = b2[fcol];
    const int nb2 = mt * 16 + ((lane & 16) ? 8 : 0);
    for (int r = 0; r < 8; ++r) {
      const int n = nb2 + r, flat = tok0 + n;
      out[(size_t)((flat >> 14) * 128 + fcol) * 16384 + (flat & 16383)] =
          sX[n * 128 + fcol] + acc[r] + bias;
    }
  }
}

// ---------------------------------------------------------------------------
extern "C" void kernel_launch(void* const* d_in, const int* in_sizes, int n_in,
                              void* d_out, int out_size, void* d_ws, size_t ws_size,
                              hipStream_t stream) {
  (void)in_sizes; (void)n_in; (void)out_size; (void)ws_size;
  const float* x     = (const float*)d_in[0];
  const float* ln1g  = (const float*)d_in[1];
  const float* ln1b  = (const float*)d_in[2];
  const float* w_ea  = (const float*)d_in[3];
  const float* w_sa  = (const float*)d_in[4];
  const float* rpb   = (const float*)d_in[5];
  const float* fc11w = (const float*)d_in[6];
  const float* fc11b = (const float*)d_in[7];
  const float* fc12w = (const float*)d_in[8];
  const float* fc12b = (const float*)d_in[9];
  const float* fc2w  = (const float*)d_in[10];
  const float* fc2b  = (const float*)d_in[11];
  const float* ln2g  = (const float*)d_in[12];
  const float* ln2b  = (const float*)d_in[13];
  const float* m11w  = (const float*)d_in[14];
  const float* m11b  = (const float*)d_in[15];
  const float* m12w  = (const float*)d_in[16];
  const float* m12b  = (const float*)d_in[17];
  const float* m2w   = (const float*)d_in[18];
  const float* m2b   = (const float*)d_in[19];

  char* ws = (char*)d_ws;
  h16*   yb = (h16*)ws;                       // 512*64*1536 f16 = 96 MB
  float* x1 = (float*)(ws + 100663296);       // 16 MB
  h16*   wf = (h16*)(ws + 117440512);         // f16 weight pool (~1 MB)
  h16* f_ea  = wf;
  h16* f_sa  = wf + 12288;
  h16* f_11  = wf + 24576;
  h16* f_12  = wf + 221184;
  h16* f_2   = wf + 417792;
  h16* f_m11 = wf + 434176;
  h16* f_m12 = wf + 450560;
  h16* f_m2  = wf + 466944;

  k_cvt<<<(12288 + 255) / 256, 256, 0, stream>>>(w_ea, f_ea, 12288);
  k_cvt<<<(12288 + 255) / 256, 256, 0, stream>>>(w_sa, f_sa, 12288);
  k_cvt<<<(196608 + 255) / 256, 256, 0, stream>>>(fc11w, f_11, 196608);
  k_cvt<<<(196608 + 255) / 256, 256, 0, stream>>>(fc12w, f_12, 196608);
  k_cvt<<<(16384 + 255) / 256, 256, 0, stream>>>(fc2w, f_2, 16384);
  k_cvt<<<(16384 + 255) / 256, 256, 0, stream>>>(m11w, f_m11, 16384);
  k_cvt<<<(16384 + 255) / 256, 256, 0, stream>>>(m12w, f_m12, 16384);
  k_cvt<<<(16384 + 255) / 256, 256, 0, stream>>>(m2w, f_m2, 16384);

  k_attn<<<512, 256, 282624, stream>>>(x, ln1g, ln1b, f_ea, f_sa, rpb, yb);
  k_geglu1<<<512, 256, 278528, stream>>>(yb, f_11, f_12, fc11b, fc12b, f_2, fc2b,
                                         x, x1);
  k_mlp2<<<512, 256, 131072, stream>>>(x1, ln2g, ln2b, f_m11, m11b, f_m12, m12b,
                                       f_m2, m2b, (float*)d_out);
}
